// MultiHeadedAttention_38869454029277
// MI455X (gfx1250) — matrix operand order repaired and checked by reading, not compiled, not run on hardware
//
#include <hip/hip_runtime.h>
#include <hip/hip_bf16.h>

// MHA: B=4, L=2048, DIM=768, H=12, DK=64.
// cvt_x / cvt_wt : one-time f32->bf16 conversion (+weight transpose)
// qkv_gemm      : Y = Xb @ WT^T + b  -> Q,K [B,H,L,DK], V transposed [B,H,DK,L]
//                 (staging: global_load_async_to_lds_b128, double-buffered)
// attention     : flash attention, bf16 WMMA, online softmax -> Xa bf16
//                 (staging: TDM tensor_load_to_lds, double-buffered, TENSORcnt)
// out_proj      : Out = Xa @ WoT^T + bo (f32)
// Workspace: 5*6291456 + 4*589824 bf16 elems ~= 67.6 MB.

typedef __attribute__((ext_vector_type(16))) __bf16 bf16x16;
typedef __attribute__((ext_vector_type(8)))  __bf16 bf16x8;
typedef __attribute__((ext_vector_type(8)))  float  f32x8;
typedef __attribute__((ext_vector_type(4)))  unsigned u32x4;
typedef __attribute__((ext_vector_type(8)))  unsigned u32x8;

__device__ inline bf16x16 make_frag(const __bf16* p0, const __bf16* p1) {
  union { bf16x16 v; bf16x8 h[2]; } f;
  f.h[0] = *(const bf16x8*)p0;
  f.h[1] = *(const bf16x8*)p1;
  return f.v;
}

__device__ inline f32x8 wmma_bf16(bf16x16 a, bf16x16 b, f32x8 c) {
  return __builtin_amdgcn_wmma_f32_16x16x32_bf16(false, a, false, b, (short)0, c,
                                                 false, false);
}

// Async 16B global->LDS copy (ASYNCcnt-tracked).
__device__ inline void async_b128(const void* gbase, unsigned byteOff, void* ldsPtr) {
  unsigned lds = (unsigned)(unsigned long long)ldsPtr;   // low 32 bits = LDS addr
  asm volatile("global_load_async_to_lds_b128 %0, %1, %2"
               :: "v"(lds), "v"(byteOff), "s"(gbase) : "memory");
}
__device__ inline void wait_async0() { asm volatile("s_wait_asynccnt 0x0" ::: "memory"); }
__device__ inline void wait_async2() { asm volatile("s_wait_asynccnt 0x2" ::: "memory"); }

// TDM descriptor group 0: count=1, lds byte addr, 57-bit global addr, type=2.
__device__ inline u32x4 tdm_g0(void* ldsPtr, unsigned long long ga) {
  u32x4 g;
  g[0] = 1u;                                        // count = 1 (valid D#)
  g[1] = (unsigned)(unsigned long long)ldsPtr;      // lds_addr
  g[2] = (unsigned)ga;                              // global_addr[31:0]
  g[3] = (unsigned)((ga >> 32) & 0x01FFFFFFull)     // global_addr[56:32]
       | (2u << 30);                                // type = 2 ("image")
  return g;
}

// 2D tensor_load_to_lds: tile packed contiguously into LDS.
__device__ inline void tdm_load(u32x4 g0, u32x8 g1) {
  asm volatile("tensor_load_to_lds %0, %1" :: "s"(g0), "s"(g1) : "memory");
}

// ---------------------------------------------------------------------------
// One-time converts.
// ---------------------------------------------------------------------------
__global__ void __launch_bounds__(256) cvt_x_kernel(const float* __restrict__ X,
                                                    __bf16* __restrict__ Xb) {
  size_t i = ((size_t)blockIdx.x * 256 + threadIdx.x) * 8;
  const float4* s = (const float4*)(X + i);
  float4 a = s[0], b = s[1];
  bf16x8 o;
  o[0] = (__bf16)a.x; o[1] = (__bf16)a.y; o[2] = (__bf16)a.z; o[3] = (__bf16)a.w;
  o[4] = (__bf16)b.x; o[5] = (__bf16)b.y; o[6] = (__bf16)b.z; o[7] = (__bf16)b.w;
  *(bf16x8*)(Xb + i) = o;
}

// WT[n][k] = (bf16)W[k][n]; 32x32 LDS-tiled transpose. grid=(24,24,4 weights).
__global__ void __launch_bounds__(256) cvt_wt_kernel(
    const float* __restrict__ Wq, const float* __restrict__ Wk,
    const float* __restrict__ Wv, const float* __restrict__ Wo,
    __bf16* __restrict__ Tq, __bf16* __restrict__ Tk,
    __bf16* __restrict__ Tv, __bf16* __restrict__ To) {
  const float* W; __bf16* T;
  switch (blockIdx.z) {
    case 0:  W = Wq; T = Tq; break;
    case 1:  W = Wk; T = Tk; break;
    case 2:  W = Wv; T = Tv; break;
    default: W = Wo; T = To; break;
  }
  __shared__ float sT[32][33];
  const int tid = threadIdx.x;
  const int c = tid & 31, r = tid >> 5;
  const int k0 = blockIdx.x * 32, n0 = blockIdx.y * 32;
  #pragma unroll
  for (int i = 0; i < 4; ++i)
    sT[r + 8 * i][c] = W[(size_t)(k0 + r + 8 * i) * 768 + n0 + c];
  __syncthreads();
  #pragma unroll
  for (int i = 0; i < 4; ++i)
    T[(size_t)(n0 + r + 8 * i) * 768 + k0 + c] = (__bf16)sT[c][r + 8 * i];
}

// ---------------------------------------------------------------------------
// Kernel 1: Q/K/V projections (async-LDS double buffer).
// grid = (12 n-tiles, 128 m-tiles, 3 {Q,K,V}), block = 256 (8 waves).
// ---------------------------------------------------------------------------
__global__ void __launch_bounds__(256) qkv_gemm_kernel(
    const __bf16* __restrict__ Xb,
    const __bf16* __restrict__ Tq, const __bf16* __restrict__ Tk,
    const __bf16* __restrict__ Tv,
    const float* __restrict__ bq, const float* __restrict__ bk,
    const float* __restrict__ bv,
    __bf16* __restrict__ Qo, __bf16* __restrict__ Ko, __bf16* __restrict__ VTo)
{
  const __bf16* WT; const float* bias;
  const int z = blockIdx.z;
  if (z == 0)      { WT = Tq; bias = bq; }
  else if (z == 1) { WT = Tk; bias = bk; }
  else             { WT = Tv; bias = bv; }

  __shared__ __bf16 sA[2][64][32];
  __shared__ __bf16 sB[2][64][32];

  const int tid   = threadIdx.x;
  const int lane  = tid & 31;
  const int w     = tid >> 5;
  const int mBase = blockIdx.y * 64;
  const int nBase = blockIdx.x * 64;
  const int mLoc  = (w & 3) * 16;
  const int nLoc  = (w >> 2) * 32;
  const int laneN = lane & 15;
  const int kbA   = (lane >> 4) * 16;
  const int kbB   = (lane >> 4) * 16;
  const int rowS  = tid >> 2;
  const int colS  = (tid & 3) * 8;

  auto issue = [&](int s) {
    int buf = s & 1, k0 = s * 32;
    async_b128(Xb, (unsigned)(((mBase + rowS) * 768 + k0 + colS) * 2),
               &sA[buf][rowS][colS]);
    async_b128(WT, (unsigned)(((nBase + rowS) * 768 + k0 + colS) * 2),
               &sB[buf][rowS][colS]);
  };

  f32x8 acc0 = {}; f32x8 acc1 = {};
  issue(0);
  for (int s = 0; s < 24; ++s) {
    __syncthreads();
    if (s < 23) { issue(s + 1); wait_async2(); } else { wait_async0(); }
    __syncthreads();
    const int buf = s & 1;
    bf16x16 a  = make_frag(&sA[buf][mLoc + laneN][kbA],
                           &sA[buf][mLoc + laneN][kbA + 8]);
    bf16x16 b0 = make_frag(&sB[buf][nLoc + laneN][kbB],
                           &sB[buf][nLoc + laneN][kbB + 8]);
    bf16x16 b1 = make_frag(&sB[buf][nLoc + 16 + laneN][kbB],
                           &sB[buf][nLoc + 16 + laneN][kbB + 8]);
    acc0 = wmma_bf16(a, b0, acc0);
    acc1 = wmma_bf16(a, b1, acc1);
  }

  const int rOff = (lane >> 4) * 8;
  #pragma unroll
  for (int nb = 0; nb < 2; ++nb) {
    f32x8 acc = nb ? acc1 : acc0;
    int n = nBase + nLoc + nb * 16 + laneN;
    float bb = bias[n];
    int h = n >> 6, d = n & 63;
    #pragma unroll
    for (int r = 0; r < 8; ++r) {
      int m  = mBase + mLoc + rOff + r;
      int bi = m >> 11, l = m & 2047;
      __bf16 v = (__bf16)(acc[r] + bb);
      if (z == 2)
        VTo[(((size_t)bi * 12 + h) * 64 + d) * 2048 + l] = v;
      else if (z == 1)
        Ko[(((size_t)bi * 12 + h) * 2048 + l) * 64 + d] = v;
      else
        Qo[(((size_t)bi * 12 + h) * 2048 + l) * 64 + d] = v;
    }
  }
}

// ---------------------------------------------------------------------------
// Kernel 2: flash attention; K/V tiles staged by the Tensor Data Mover.
// grid = (16 q-tiles, 48 bh), block = 256 (8 waves x 16 q rows).
// ---------------------------------------------------------------------------
__global__ void __launch_bounds__(256) attention_kernel(
    const __bf16* __restrict__ Q, const __bf16* __restrict__ K,
    const __bf16* __restrict__ VT, const int* __restrict__ mask,
    __bf16* __restrict__ X)
{
  __shared__ __bf16 sK[2][32][64];    // [key][d]   (4 KB tile, packed)
  __shared__ __bf16 sVT[2][64][32];   // [d][key]   (4 KB tile, packed)
  __shared__ __bf16 sP[8][16][32];    // wave-private P (C->A relayout)

  const int tid   = threadIdx.x;
  const int lane  = tid & 31;
  const int w     = tid >> 5;
  const int bh    = blockIdx.y;
  const int b     = bh / 12, h = bh % 12;
  const int qBase = blockIdx.x * 128 + w * 16;
  const int laneN = lane & 15;
  const int kbA   = (lane >> 4) * 16;
  const int kbB   = (lane >> 4) * 16;
  const int rOff  = (lane >> 4) * 8;

  // TDM descriptors. Group1 is loop-invariant.
  // K tile:  tensor [2048 keys x 64 d], tile 32x64, dim0_stride=64.
  // V tile:  tensor [64 d x 2048 keys], tile 64x32, dim0_stride=2048.
  const u32x8 g1K = { (1u << 16),        // data_size = 2 bytes
                      (64u << 16),       // tensor_dim0 = 64        (bits 79:48)
                      (2048u << 16),     // tensor_dim1 = 2048      (bits 111:80)
                      (64u << 16),       // tile_dim0 = 64          (bits 127:112)
                      32u,               // tile_dim1 = 32, tile_dim2 = 0
                      64u, 0u, 0u };     // tensor_dim0_stride = 64
  const u32x8 g1V = { (1u << 16),
                      (2048u << 16),     // tensor_dim0 = 2048
                      (64u << 16),       // tensor_dim1 = 64
                      (32u << 16),       // tile_dim0 = 32
                      64u,               // tile_dim1 = 64
                      2048u, 0u, 0u };   // tensor_dim0_stride = 2048
  const unsigned long long kBaseGA =
      (unsigned long long)K  + (unsigned long long)bh * 2048ull * 64ull * 2ull;
  const unsigned long long vBaseGA =
      (unsigned long long)VT + (unsigned long long)bh * 64ull * 2048ull * 2ull;

  auto issue = [&](int t) {
    if (w == 0) {                        // one wave drives the TDM
      int buf = t & 1, kblk = t * 32;
      tdm_load(tdm_g0(&sK[buf][0][0],  kBaseGA + (unsigned long long)kblk * 64 * 2), g1K);
      tdm_load(tdm_g0(&sVT[buf][0][0], vBaseGA + (unsigned long long)kblk * 2),      g1V);
    }
  };

  // Q fragments pinned in VGPRs for the whole pass
  const __bf16* qrow = Q + ((size_t)bh * 2048 + qBase + laneN) * 64;
  bf16x16 qa0 = make_frag(qrow + kbA,      qrow + kbA + 8);
  bf16x16 qa1 = make_frag(qrow + 32 + kbA, qrow + 32 + kbA + 8);

  f32x8 o0 = {}, o1 = {}, o2 = {}, o3 = {};
  float mrow[8], lrow[8];
  #pragma unroll
  for (int r = 0; r < 8; ++r) { mrow[r] = -3.0e38f; lrow[r] = 0.f; }
  const float scale = 0.125f;   // 1/sqrt(64)
  const int* mrowp = mask + b * 2048;

  issue(0);
  for (int t = 0; t < 64; ++t) {
    __syncthreads();                     // all waves done reading buf (t+1)&1
    if (t < 63) {
      issue(t + 1);
      if (w == 0) __builtin_amdgcn_s_wait_tensorcnt(2);   // drain tile t
    } else {
      if (w == 0) __builtin_amdgcn_s_wait_tensorcnt(0);
    }
    __syncthreads();                     // publish tile t to all waves
    const int buf = t & 1, kblk = t * 32;

    // scores: 16q x 32keys
    f32x8 s0 = {}, s1 = {};
    bf16x16 bk0  = make_frag(&sK[buf][laneN][kbB],           &sK[buf][laneN][kbB + 8]);
    bf16x16 bk0h = make_frag(&sK[buf][laneN][32 + kbB],      &sK[buf][laneN][32 + kbB + 8]);
    bf16x16 bk1  = make_frag(&sK[buf][16 + laneN][kbB],      &sK[buf][16 + laneN][kbB + 8]);
    bf16x16 bk1h = make_frag(&sK[buf][16 + laneN][32 + kbB], &sK[buf][16 + laneN][32 + kbB + 8]);
    s0 = wmma_bf16(qa0, bk0, s0);  s0 = wmma_bf16(qa1, bk0h, s0);
    s1 = wmma_bf16(qa0, bk1, s1);  s1 = wmma_bf16(qa1, bk1h, s1);

    const int m0 = mrowp[kblk + laneN];
    const int m1 = mrowp[kblk + 16 + laneN];

    #pragma unroll
    for (int r = 0; r < 8; ++r) {
      float v0 = m0 ? s0[r] * scale : -1.0e9f;
      float v1 = m1 ? s1[r] * scale : -1.0e9f;
      float pm = fmaxf(v0, v1);
      #pragma unroll
      for (int i = 1; i < 16; i <<= 1) pm = fmaxf(pm, __shfl_xor(pm, i, 32));
      float mnew  = fmaxf(mrow[r], pm);
      float alpha = __expf(mrow[r] - mnew);
      float p0 = __expf(v0 - mnew);
      float p1 = __expf(v1 - mnew);
      float ps = p0 + p1;
      #pragma unroll
      for (int i = 1; i < 16; i <<= 1) ps += __shfl_xor(ps, i, 32);
      lrow[r] = lrow[r] * alpha + ps;
      mrow[r] = mnew;
      o0[r] *= alpha; o1[r] *= alpha; o2[r] *= alpha; o3[r] *= alpha;
      sP[w][rOff + r][laneN]      = (__bf16)p0;
      sP[w][rOff + r][16 + laneN] = (__bf16)p1;
    }
    // wave-private LDS relayout: DS ops in-order per wave; drain + compiler barrier
    asm volatile("s_wait_dscnt 0x0" ::: "memory");

    bf16x16 pa  = make_frag(&sP[w][laneN][kbA],         &sP[w][laneN][kbA + 8]);
    bf16x16 vf0 = make_frag(&sVT[buf][laneN][kbB],      &sVT[buf][laneN][kbB + 8]);
    bf16x16 vf1 = make_frag(&sVT[buf][16 + laneN][kbB], &sVT[buf][16 + laneN][kbB + 8]);
    bf16x16 vf2 = make_frag(&sVT[buf][32 + laneN][kbB], &sVT[buf][32 + laneN][kbB + 8]);
    bf16x16 vf3 = make_frag(&sVT[buf][48 + laneN][kbB], &sVT[buf][48 + laneN][kbB + 8]);
    o0 = wmma_bf16(pa, vf0, o0);
    o1 = wmma_bf16(pa, vf1, o1);
    o2 = wmma_bf16(pa, vf2, o2);
    o3 = wmma_bf16(pa, vf3, o3);
  }

  #pragma unroll
  for (int r = 0; r < 8; ++r) {
    float inv = 1.0f / lrow[r];
    int l = qBase + rOff + r;
    __bf16* xp = X + ((size_t)b * 2048 + l) * 768 + h * 64;
    xp[laneN]      = (__bf16)(o0[r] * inv);
    xp[16 + laneN] = (__bf16)(o1[r] * inv);
    xp[32 + laneN] = (__bf16)(o2[r] * inv);
    xp[48 + laneN] = (__bf16)(o3[r] * inv);
  }
}

// ---------------------------------------------------------------------------
// Kernel 3: Out = Xa @ WoT^T + bo (f32), async-LDS double buffer.
// grid = (12, 128), block = 256.
// ---------------------------------------------------------------------------
__global__ void __launch_bounds__(256) out_proj_kernel(
    const __bf16* __restrict__ Xa, const __bf16* __restrict__ To,
    const float* __restrict__ bo, float* __restrict__ Out)
{
  __shared__ __bf16 sA[2][64][32];
  __shared__ __bf16 sB[2][64][32];

  const int tid   = threadIdx.x;
  const int lane  = tid & 31;
  const int w     = tid >> 5;
  const int mBase = blockIdx.y * 64;
  const int nBase = blockIdx.x * 64;
  const int mLoc  = (w & 3) * 16;
  const int nLoc  = (w >> 2) * 32;
  const int laneN = lane & 15;
  const int kbA   = (lane >> 4) * 16;
  const int kbB   = (lane >> 4) * 16;
  const int rowS  = tid >> 2;
  const int colS  = (tid & 3) * 8;

  auto issue = [&](int s) {
    int buf = s & 1, k0 = s * 32;
    async_b128(Xa, (unsigned)(((mBase + rowS) * 768 + k0 + colS) * 2),
               &sA[buf][rowS][colS]);
    async_b128(To, (unsigned)(((nBase + rowS) * 768 + k0 + colS) * 2),
               &sB[buf][rowS][colS]);
  };

  f32x8 acc0 = {}; f32x8 acc1 = {};
  issue(0);
  for (int s = 0; s < 24; ++s) {
    __syncthreads();
    if (s < 23) { issue(s + 1); wait_async2(); } else { wait_async0(); }
    __syncthreads();
    const int buf = s & 1;
    bf16x16 a  = make_frag(&sA[buf][mLoc + laneN][kbA],
                           &sA[buf][mLoc + laneN][kbA + 8]);
    bf16x16 b0 = make_frag(&sB[buf][nLoc + laneN][kbB],
                           &sB[buf][nLoc + laneN][kbB + 8]);
    bf16x16 b1 = make_frag(&sB[buf][nLoc + 16 + laneN][kbB],
                           &sB[buf][nLoc + 16 + laneN][kbB + 8]);
    acc0 = wmma_bf16(a, b0, acc0);
    acc1 = wmma_bf16(a, b1, acc1);
  }

  const int rOff = (lane >> 4) * 8;
  #pragma unroll
  for (int nb = 0; nb < 2; ++nb) {
    f32x8 acc = nb ? acc1 : acc0;
    int n = nBase + nLoc + nb * 16 + laneN;
    float bb = bo[n];
    #pragma unroll
    for (int r = 0; r < 8; ++r) {
      int m = mBase + mLoc + rOff + r;
      Out[(size_t)m * 768 + n] = acc[r] + bb;
    }
  }
}

// ---------------------------------------------------------------------------
extern "C" void kernel_launch(void* const* d_in, const int* in_sizes, int n_in,
                              void* d_out, int out_size, void* d_ws, size_t ws_size,
                              hipStream_t stream) {
  const float* qkv = (const float*)d_in[0];
  const int*   msk = (const int*)d_in[1];
  const float* Wq  = (const float*)d_in[2];
  const float* bq  = (const float*)d_in[3];
  const float* Wk  = (const float*)d_in[4];
  const float* bk  = (const float*)d_in[5];
  const float* Wv  = (const float*)d_in[6];
  const float* bv  = (const float*)d_in[7];
  const float* Wo  = (const float*)d_in[8];
  const float* bo  = (const float*)d_in[9];
  float* out = (float*)d_out;

  const size_t n  = (size_t)4 * 12 * 2048 * 64;   // 6,291,456 (== 8192*768)
  const size_t nW = (size_t)768 * 768;            // 589,824
  __bf16* ws  = (__bf16*)d_ws;                    // needs ~67.6 MB
  __bf16* Qb  = ws;
  __bf16* Kb  = ws + n;
  __bf16* VTb = ws + 2 * n;
  __bf16* Xa  = ws + 3 * n;
  __bf16* Xqb = ws + 4 * n;
  __bf16* Tq  = ws + 5 * n;
  __bf16* Tk  = Tq + nW;
  __bf16* Tv  = Tk + nW;
  __bf16* To  = Tv + nW;

  cvt_x_kernel<<<dim3(3072), 256, 0, stream>>>(qkv, Xqb);
  cvt_wt_kernel<<<dim3(24, 24, 4), 256, 0, stream>>>(Wq, Wk, Wv, Wo,
                                                     Tq, Tk, Tv, To);
  qkv_gemm_kernel<<<dim3(12, 128, 3), 256, 0, stream>>>(Xqb, Tq, Tk, Tv,
                                                        bq, bk, bv, Qb, Kb, VTb);
  attention_kernel<<<dim3(16, 48), 256, 0, stream>>>(Qb, Kb, VTb, msk, Xa);
  out_proj_kernel<<<dim3(12, 128), 256, 0, stream>>>(Xa, To, bo, out);
}
